// AtomTransformerBlock_59940563583354
// MI455X (gfx1250) — matrix operand order repaired and checked by reading, not compiled, not run on hardware
//
#include <hip/hip_runtime.h>
#include <hip/hip_bf16.h>
#include <math.h>

typedef __bf16 bf16;
typedef __attribute__((ext_vector_type(16))) __bf16 bf16x16;
typedef __attribute__((ext_vector_type(8)))  __bf16 bf16x8;
typedef __attribute__((ext_vector_type(4)))  __bf16 bf16x4;
typedef __attribute__((ext_vector_type(8)))  float  f32x8;
typedef __attribute__((ext_vector_type(4)))  unsigned int u32x4;
typedef __attribute__((ext_vector_type(4)))  int i32x4v;
typedef __attribute__((ext_vector_type(8)))  int i32x8v;

#define DEV static __device__ __forceinline__

// Tensor Data Mover availability (device pass only; host pass uses fallback path)
#if defined(__AMDGCN__) && __has_builtin(__builtin_amdgcn_tensor_load_to_lds) && \
    __has_builtin(__builtin_amdgcn_s_wait_tensorcnt)
#define HAVE_TDM 1
#else
#define HAVE_TDM 0
#endif
#if HAVE_TDM && __has_include(<hip/amd_detail/amd_gfx1250_TDM.h>)
#define TDM_SIX_ARG 1
#else
#define TDM_SIX_ARG 0
#endif

static constexpr int B_ = 2, N_ATOM = 4096, N_RES = 512;
static constexpr int C_ATOM = 128, C_S = 256, H_ = 4, CH = 32;
static constexpr int NATOMS = B_ * N_ATOM;   // 8192
static constexpr int NRES   = B_ * N_RES;    // 1024

DEV int lane_id() { return threadIdx.x & 31; }

// ---------------- WMMA bf16 16x16x32 fragment helpers (k = 16*(lane/16) + e in A and B) -----------
DEV bf16x16 load_a_frag(const bf16* s, int ld) {
  int l = lane_id(); int m = l & 15; int kh = (l >> 4) << 4;
  const bf16* p = s + m * ld;
  bf16x16 a;
#pragma unroll
  for (int e = 0; e < 8; ++e) a[e] = p[kh + e];
#pragma unroll
  for (int e = 0; e < 8; ++e) a[8 + e] = p[kh + 8 + e];
  return a;
}
DEV bf16x16 load_b_frag(const bf16* s, int ld) {   // B KxN row-major
  int l = lane_id(); int n = l & 15; int kb = (l >> 4) << 4;
  bf16x16 b;
#pragma unroll
  for (int e = 0; e < 16; ++e) b[e] = s[(kb + e) * ld + n];
  return b;
}
DEV bf16x16 load_bT_frag(const bf16* s, int ld) {  // B = S^T, S is NxK row-major
  int l = lane_id(); int n = l & 15; int kb = (l >> 4) << 4;
  const bf16* p = s + n * ld + kb;
  bf16x16 b;
#pragma unroll
  for (int e = 0; e < 16; ++e) b[e] = p[e];
  return b;
}
DEV f32x8 wmma_bf16(bf16x16 a, bf16x16 b, f32x8 c) {
  return __builtin_amdgcn_wmma_f32_16x16x32_bf16(false, a, false, b, (short)0, c, false, false);
}
DEV float sigmoidf_(float x) { return 1.f / (1.f + __expf(-x)); }
DEV float wave_sum(float v) {
#pragma unroll
  for (int off = 16; off > 0; off >>= 1) v += __shfl_xor(v, off, 32);
  return v;
}

#if HAVE_TDM
// Issue a TDM DMA of a 2-D bf16 tile (tile_rows x tile_cols) from global into LDS.
// Rows >= tensor_rows are zero-filled by TDM OOB semantics (used for causal masking).
// D# layout per CDNA5 ISA section 8: group0 = {flags, lds_addr, global_addr, type=2},
// group1 = {data_size=2B, tensor_dim0/1, tile_dim0/1, tensor_dim0_stride}.
DEV void tdm_load_2d_bf16(const void* gsrc, void* ldst, int tile_cols, int tile_rows,
                          int tensor_rows, int row_stride_elems) {
  unsigned long long ga = (unsigned long long)(uintptr_t)gsrc;
  unsigned int la = (unsigned int)(uintptr_t)ldst;  // low 32 bits of flat addr = LDS offset
  u32x4 g0;
  g0[0] = 1u;                                                   // count=1 (valid user D#)
  g0[1] = la;                                                   // lds_addr [63:32]
  g0[2] = (unsigned int)ga;                                     // global_addr [95:64]
  g0[3] = (unsigned int)((ga >> 32) & 0x1FFFFFFull) | (2u << 30); // ga[56:32] + type=2
  unsigned int td0 = (unsigned int)tile_cols;
  unsigned int td1 = (unsigned int)tensor_rows;
  i32x8v g1;
  g1[0] = (int)(1u << 16);                                      // data_size=1 -> 2 bytes
  g1[1] = (int)(td0 << 16);                                     // tensor_dim0[15:0] @ 63:48
  g1[2] = (int)((td0 >> 16) | (td1 << 16));                     // td0 hi | td1 lo
  g1[3] = (int)((td1 >> 16) | ((unsigned int)tile_cols << 16)); // td1 hi | tile_dim0
  g1[4] = (int)((unsigned int)tile_rows & 0xFFFFu);             // tile_dim1 (tile_dim2=0)
  g1[5] = (int)(unsigned int)row_stride_elems;                  // tensor_dim0_stride lo
  g1[6] = 0;
  g1[7] = 0;
  i32x4v z4 = {0, 0, 0, 0};
#if TDM_SIX_ARG
  i32x8v z8 = {0, 0, 0, 0, 0, 0, 0, 0};
  __builtin_amdgcn_tensor_load_to_lds(g0, g1, z4, z4, z8, 0);
#else
  __builtin_amdgcn_tensor_load_to_lds(g0, g1, z4, z4, 0);
#endif
}
#endif  // HAVE_TDM

// ---------------- weight pack: fp32 -> bf16 with optional per-row scale -----------------
__global__ void pack_w_kernel(const float* __restrict__ W, const float* __restrict__ scale,
                              bf16* __restrict__ dst, int K, int Nsub, int stride, int colOff) {
  int i = blockIdx.x * 256 + threadIdx.x;
  if (i >= K * Nsub) return;
  int k = i / Nsub, n = i - k * Nsub;
  float s = scale ? scale[k] : 1.f;
  dst[(size_t)k * stride + colOff + n] = (bf16)(W[i] * s);
}

// ---------------- residue LN prep ------------------------------------------------------
__global__ __launch_bounds__(256) void res_prep_kernel(const float* __restrict__ res,
                                                       bf16* __restrict__ lnout,
                                                       bf16* __restrict__ rawout) {
  int row = blockIdx.x * 8 + (threadIdx.x >> 5);
  int l = lane_id();
  const float* r = res + (size_t)row * C_S;
  float v[8]; float s = 0.f;
#pragma unroll
  for (int i = 0; i < 8; ++i) { v[i] = r[l + i * 32]; s += v[i]; }
  float mu = wave_sum(s) * (1.f / C_S);
  float vs = 0.f;
#pragma unroll
  for (int i = 0; i < 8; ++i) { float d = v[i] - mu; vs += d * d; }
  float rs = rsqrtf(wave_sum(vs) * (1.f / C_S) + 1e-5f);
#pragma unroll
  for (int i = 0; i < 8; ++i) {
    int c = l + i * 32;
    lnout[(size_t)row * C_S + c]  = (bf16)((v[i] - mu) * rs);
    rawout[(size_t)row * C_S + c] = (bf16)v[i];
  }
}

// ---------------- adaLN over atoms -----------------------------------------------------
__global__ __launch_bounds__(256) void adaln_kernel(const float* __restrict__ x,
                                                    const float* __restrict__ gate,
                                                    const float* __restrict__ bias,
                                                    const int* __restrict__ res_idx,
                                                    bf16* __restrict__ out) {
  int row = blockIdx.x * 8 + (threadIdx.x >> 5);
  int l = lane_id();
  int rr = res_idx[row] + (row / N_ATOM) * N_RES;
  const float* xr = x + (size_t)row * C_ATOM;
  float v[4]; float s = 0.f;
#pragma unroll
  for (int i = 0; i < 4; ++i) { v[i] = xr[l + i * 32]; s += v[i]; }
  float mu = wave_sum(s) * (1.f / C_ATOM);
  float vs = 0.f;
#pragma unroll
  for (int i = 0; i < 4; ++i) { float d = v[i] - mu; vs += d * d; }
  float rs = rsqrtf(wave_sum(vs) * (1.f / C_ATOM) + 1e-5f);
#pragma unroll
  for (int i = 0; i < 4; ++i) {
    int c = l + i * 32;
    float y = (v[i] - mu) * rs;
    out[(size_t)row * C_ATOM + c] =
        (bf16)(y * gate[(size_t)rr * C_ATOM + c] + bias[(size_t)rr * C_ATOM + c]);
  }
}

// ---------------- bf16 WMMA GEMM: C(f32) = A @ B, compile-time shapes -------------------
// block = 256 threads (8 waves), BM=64 BN=128 BK=32; grid = (M/64, N/128).
// B k-slab staged by the Tensor Data Mover (TDM) when available.
template <int M, int N, int K>
__global__ __launch_bounds__(256) void gemm_bf16_kernel(const bf16* __restrict__ A,
                                                        const bf16* __restrict__ B,
                                                        float* __restrict__ C) {
  __shared__ __attribute__((aligned(16))) bf16 As[64 * 32];
  __shared__ __attribute__((aligned(16))) bf16 Bs[32 * 128];
  const int m0 = blockIdx.x * 64;
  const int n0 = blockIdx.y * 128;
  const int tid = threadIdx.x;
  const int wave = tid >> 5;
  f32x8 acc[4] = {};
  for (int k0 = 0; k0 < K; k0 += 32) {
    __syncthreads();
    {  // stage A 64x32 (8 elems / thread, 16B vectors)
      int idx = tid * 8;
      int r = idx >> 5, c = idx & 31;
      *(bf16x8*)&As[idx] = *(const bf16x8*)&A[(size_t)(m0 + r) * K + k0 + c];
    }
#if HAVE_TDM
    if (wave == 0) {  // DMA B 32x128 slab into LDS via TDM
      tdm_load_2d_bf16(&B[(size_t)k0 * N + n0], Bs, 128, 32, 32, N);
      __builtin_amdgcn_s_wait_tensorcnt(0);
    }
#else
    {  // fallback: manual stage B 32x128
      int idx = tid * 16;
      int r = idx >> 7, c = idx & 127;
      *(bf16x8*)&Bs[idx]     = *(const bf16x8*)&B[(size_t)(k0 + r) * N + n0 + c];
      *(bf16x8*)&Bs[idx + 8] = *(const bf16x8*)&B[(size_t)(k0 + r) * N + n0 + c + 8];
    }
#endif
    __syncthreads();
    bf16x16 b = load_b_frag(Bs + wave * 16, 128);
#pragma unroll
    for (int t = 0; t < 4; ++t) {
      bf16x16 a = load_a_frag(As + t * 16 * 32, 32);
      acc[t] = wmma_bf16(a, b, acc[t]);
    }
  }
  const int l = lane_id();
  const int n = l & 15, mb = (l >> 4) << 3;
  const int cn = n0 + wave * 16 + n;
#pragma unroll
  for (int t = 0; t < 4; ++t)
#pragma unroll
    for (int r = 0; r < 8; ++r)
      C[(size_t)(m0 + t * 16 + mb + r) * N + cn] = acc[t][r];
}

// ---------------- epilogues ------------------------------------------------------------
__global__ void res_epilogue_kernel(const float* __restrict__ raw1, const float* __restrict__ raw2,
                                    const float* __restrict__ gate_b, const float* __restrict__ tgate_b,
                                    const float* __restrict__ sgate_b, const float* __restrict__ tcond_b,
                                    float* g1, float* b1, float* g2, float* b2, float* sg, float* tg) {
  int i = blockIdx.x * 256 + threadIdx.x;  // NRES*128
  int row = i >> 7, c = i & 127;
  const float* r1 = raw1 + (size_t)row * 512;
  const float* r2 = raw2 + (size_t)row * 256;
  g1[i] = sigmoidf_(r1[c] + gate_b[c]);
  b1[i] = r1[128 + c];
  g2[i] = sigmoidf_(r1[256 + c] + tgate_b[c]);
  b2[i] = r1[384 + c];
  sg[i] = sigmoidf_(r2[c] + sgate_b[c]);
  tg[i] = sigmoidf_(r2[128 + c] + tcond_b[c]);
}

__global__ void qkvg_epilogue_kernel(const float* __restrict__ raw, const float* __restrict__ q_b,
                                     const float* __restrict__ g_b,
                                     bf16* qso, bf16* kso, bf16* vso, float* sig_g) {
  int i = blockIdx.x * 256 + threadIdx.x;  // NATOMS*128
  int row = i >> 7, c = i & 127;
  const float* r = raw + (size_t)row * 512;
  qso[i] = (bf16)((r[c] + q_b[c]) * 0.17677669529663687f);  // 1/sqrt(CH)
  kso[i] = (bf16)r[128 + c];
  vso[i] = (bf16)r[256 + c];
  sig_g[i] = sigmoidf_(r[384 + c] + g_b[c]);
}

__global__ void bmid_epilogue_kernel(const float* __restrict__ raw, bf16* __restrict__ bmid) {
  int i = blockIdx.x * 256 + threadIdx.x;  // NATOMS*256
  int row = i >> 8, c = i & 255;
  const float* r = raw + (size_t)row * 512;
  float a = r[c], hh = r[256 + c];
  bmid[i] = (bf16)(a * sigmoidf_(a) * hh);  // silu(a)*h
}

__global__ void final_epilogue_kernel(const float* __restrict__ x, const float* __restrict__ rawY,
                                      const float* __restrict__ tg, const int* __restrict__ res_idx,
                                      float* __restrict__ out) {
  int i = blockIdx.x * 256 + threadIdx.x;  // NATOMS*128
  int row = i >> 7, c = i & 127;
  int rr = res_idx[row] + (row / N_ATOM) * N_RES;
  out[i] = x[i] + tg[(size_t)rr * C_ATOM + c] * rawY[i];
}

// ---------------- block-local attention ------------------------------------------------
// grid = (B*128 dst-blocks, 4 heads), block = 256 threads. Dst block j attends to the
// 32*min(j+1,4) atoms in blocks max(0,j-3)..j (analytic form of edge_index).
// Q/K/V head slices staged by TDM; K/V rows >= S zero-filled by TDM OOB semantics.
__global__ __launch_bounds__(256) void attn_kernel(
    const bf16* __restrict__ qs, const bf16* __restrict__ ks, const bf16* __restrict__ vs,
    const float* __restrict__ sig_g, const float* __restrict__ sg,
    const float* __restrict__ atomf, const float* __restrict__ pos,
    const float* __restrict__ rot, const float* __restrict__ trans,
    const int* __restrict__ res_idx,
    const float* __restrict__ dvW, const float* __restrict__ dW,
    const float* __restrict__ lnw, const float* __restrict__ lnb,
    const float* __restrict__ bijW, float* __restrict__ xout) {
  const int bb = blockIdx.x;
  const int h  = blockIdx.y;
  const int b  = bb >> 7, j = bb & 127;
  const int jlo = (j >= 3) ? (j - 3) : 0;
  const int S   = (j - jlo + 1) * 32;
  const int gd0 = b * N_ATOM + j * 32;
  const int gs0 = b * N_ATOM + jlo * 32;
  const int tid = threadIdx.x, wave = tid >> 5, l = tid & 31;

  __shared__ __attribute__((aligned(16))) bf16 Ql[32 * CH];
  __shared__ __attribute__((aligned(16))) bf16 Kl[128 * CH];
  __shared__ __attribute__((aligned(16))) bf16 Vl[128 * CH];
  __shared__ __attribute__((aligned(16))) float Lsm[32 * 128];

#if HAVE_TDM
  if (wave == 0) {
    tdm_load_2d_bf16(&qs[(size_t)gd0 * C_ATOM + h * CH], Ql, CH, 32, 32, C_ATOM);
    tdm_load_2d_bf16(&ks[(size_t)gs0 * C_ATOM + h * CH], Kl, CH, 128, S, C_ATOM);
    tdm_load_2d_bf16(&vs[(size_t)gs0 * C_ATOM + h * CH], Vl, CH, 128, S, C_ATOM);
    __builtin_amdgcn_s_wait_tensorcnt(0);
  }
#else
  {  // stage Q (head slice)
    int idx = tid * 4;
    int r = idx >> 5, c = idx & 31;
    *(bf16x4*)&Ql[idx] = *(const bf16x4*)&qs[(size_t)(gd0 + r) * C_ATOM + h * CH + c];
  }
  {  // stage K,V; zero-fill rows >= S
    int idx = tid * 16;
    int r = idx >> 5, c = idx & 31;
    if (r < S) {
      size_t ko = (size_t)(gs0 + r) * C_ATOM + h * CH + c;
      *(bf16x8*)&Kl[idx]     = *(const bf16x8*)&ks[ko];
      *(bf16x8*)&Kl[idx + 8] = *(const bf16x8*)&ks[ko + 8];
      *(bf16x8*)&Vl[idx]     = *(const bf16x8*)&vs[ko];
      *(bf16x8*)&Vl[idx + 8] = *(const bf16x8*)&vs[ko + 8];
    } else {
      bf16x8 z = {};
      *(bf16x8*)&Kl[idx] = z; *(bf16x8*)&Kl[idx + 8] = z;
      *(bf16x8*)&Vl[idx] = z; *(bf16x8*)&Vl[idx + 8] = z;
    }
  }
#endif
  // init logits with pair bias b_ij (or -inf mask) -- independent of Ql/Kl/Vl
  for (int e = tid; e < 32 * 128; e += 256) {
    int m = e >> 7, s = e & 127;
    float val;
    if (s >= S) {
      val = -1e30f;
    } else {
      int d = gd0 + m, sa = gs0 + s;
      int rd  = res_idx[d]  + b * N_RES;
      int rsx = res_idx[sa] + b * N_RES;
      float wx = pos[sa * 3 + 0] - pos[d * 3 + 0] - trans[rd * 3 + 0];
      float wy = pos[sa * 3 + 1] - pos[d * 3 + 1] - trans[rd * 3 + 1];
      float wz = pos[sa * 3 + 2] - pos[d * 3 + 2] - trans[rd * 3 + 2];
      const float* R = rot + (size_t)rd * 9;
      float v0 = R[0] * wx + R[3] * wy + R[6] * wz;  // R^T * w
      float v1 = R[1] * wx + R[4] * wy + R[7] * wz;
      float v2 = R[2] * wx + R[5] * wy + R[8] * wz;
      float dist = sqrtf(v0 * v0 + v1 * v1 + v2 * v2);
      float same = (rsx == rd) ? 1.f : 0.f;
      float p16[16]; float mu = 0.f;
#pragma unroll
      for (int t = 0; t < 16; ++t) {
        float pv = (v0 * dvW[t] + v1 * dvW[16 + t] + v2 * dvW[32 + t] + dist * dW[t]) * same;
        p16[t] = pv; mu += pv;
      }
      mu *= (1.f / 16.f);
      float var = 0.f;
#pragma unroll
      for (int t = 0; t < 16; ++t) { float dd = p16[t] - mu; var += dd * dd; }
      float rsq = rsqrtf(var * (1.f / 16.f) + 1e-5f);
      float bij = 0.f;
#pragma unroll
      for (int t = 0; t < 16; ++t)
        bij += ((p16[t] - mu) * rsq * lnw[t] + lnb[t]) * bijW[t * 4 + h];
      val = bij;
    }
    Lsm[e] = val;
  }
  __syncthreads();

  {  // logits += Q @ K^T  (M=32, N=128, K=CH=32): 16 tiles over 8 waves
    int mt = wave & 1;
    int ntb = (wave >> 1) * 2;
    bf16x16 a = load_a_frag(Ql + mt * 16 * CH, CH);
#pragma unroll
    for (int t = 0; t < 2; ++t) {
      int nt = ntb + t;
      bf16x16 bt = load_bT_frag(Kl + nt * 16 * CH, CH);
      f32x8 acc = {};
      acc = wmma_bf16(a, bt, acc);
      int n = l & 15, mb = (l >> 4) << 3;
#pragma unroll
      for (int r = 0; r < 8; ++r)
        Lsm[(mt * 16 + mb + r) * 128 + nt * 16 + n] += acc[r];
    }
  }
  __syncthreads();

  if (tid < 32) {  // row softmax; repack probabilities as bf16 in-place
    float* Lr = Lsm + tid * 128;
    float mx = -1e30f;
    for (int k = 0; k < 128; ++k) mx = fmaxf(mx, Lr[k]);
    float sum = 0.f;
    for (int k = 0; k < 128; ++k) { float e2 = __expf(Lr[k] - mx); Lr[k] = e2; sum += e2; }
    float inv = 1.f / sum;
    bf16* Pr = (bf16*)Lr;
    for (int k = 0; k < 128; k += 2) {
      float f0 = Lr[k], f1 = Lr[k + 1];
      Pr[k]     = (bf16)(f0 * inv);
      Pr[k + 1] = (bf16)(f1 * inv);
    }
  }
  __syncthreads();

  if (wave < 4) {  // out = P @ V (M=32, N=CH=32, K=128), fused epilogue
    int mt = wave & 1, nt = wave >> 1;
    f32x8 acc = {};
#pragma unroll
    for (int ks2 = 0; ks2 < 4; ++ks2) {
      bf16x16 a  = load_a_frag((const bf16*)Lsm + (size_t)mt * 16 * 256 + ks2 * 32, 256);
      bf16x16 bv = load_b_frag(Vl + ks2 * 32 * CH + nt * 16, CH);
      acc = wmma_bf16(a, bv, acc);
    }
    int n = l & 15, mb = (l >> 4) << 3;
#pragma unroll
    for (int r = 0; r < 8; ++r) {
      int m = mt * 16 + mb + r;
      int d = gd0 + m;
      int c = h * CH + nt * 16 + n;
      int rd = res_idx[d] + b * N_RES;
      size_t idx = (size_t)d * C_ATOM + c;
      xout[idx] = atomf[idx] + acc[r] * sig_g[idx] + sg[(size_t)rd * C_ATOM + c];
    }
  }
}

// ---------------- host-side orchestration ----------------------------------------------
extern "C" void kernel_launch(void* const* d_in, const int* in_sizes, int n_in,
                              void* d_out, int out_size, void* d_ws, size_t ws_size,
                              hipStream_t stream) {
  (void)in_sizes; (void)n_in; (void)out_size; (void)ws_size;
  const float* atomf   = (const float*)d_in[0];
  const float* pos     = (const float*)d_in[1];
  const float* resf    = (const float*)d_in[2];
  const float* rotm    = (const float*)d_in[3];
  const float* transv  = (const float*)d_in[4];
  const float* a_lnw   = (const float*)d_in[5];
  const float* a_gW    = (const float*)d_in[6];
  const float* a_gb    = (const float*)d_in[7];
  const float* a_bW    = (const float*)d_in[8];
  const float* q_W     = (const float*)d_in[9];
  const float* q_b     = (const float*)d_in[10];
  const float* kv_W    = (const float*)d_in[11];
  const float* bij_lnw = (const float*)d_in[12];
  const float* bij_lnb = (const float*)d_in[13];
  const float* bij_W   = (const float*)d_in[14];
  const float* g_W     = (const float*)d_in[15];
  const float* g_b     = (const float*)d_in[16];
  const float* sg_W    = (const float*)d_in[17];
  const float* sg_b    = (const float*)d_in[18];
  const float* dv_W    = (const float*)d_in[19];
  const float* d_W     = (const float*)d_in[20];
  const float* t_lnw   = (const float*)d_in[21];
  const float* t_gW    = (const float*)d_in[22];
  const float* t_gb    = (const float*)d_in[23];
  const float* t_bW    = (const float*)d_in[24];
  const float* l1_W    = (const float*)d_in[25];
  const float* l2_W    = (const float*)d_in[26];
  const float* tc_W    = (const float*)d_in[27];
  const float* tc_b    = (const float*)d_in[28];
  const float* lb_W    = (const float*)d_in[29];
  const int*   ridx    = (const int*)d_in[30];
  // d_in[31] = edge_index: unused (block-local pattern derived analytically)

  char* wsp = (char*)d_ws;
  size_t off = 0;
  auto alloc = [&](size_t bytes) -> char* {
    size_t cur = (off + 255) & ~(size_t)255;
    off = cur + bytes;
    return wsp + cur;
  };
  bf16*  ln_res  = (bf16*)alloc((size_t)NRES * C_S * 2);
  bf16*  res_raw = (bf16*)alloc((size_t)NRES * C_S * 2);
  bf16*  BcatA   = (bf16*)alloc((size_t)256 * 512 * 2);
  bf16*  BcatB   = (bf16*)alloc((size_t)256 * 256 * 2);
  bf16*  Wqkvg   = (bf16*)alloc((size_t)128 * 512 * 2);
  bf16*  Wlin12  = (bf16*)alloc((size_t)128 * 512 * 2);
  bf16*  Wlinb   = (bf16*)alloc((size_t)256 * 128 * 2);
  float* g1      = (float*)alloc((size_t)NRES * 128 * 4);
  float* b1      = (float*)alloc((size_t)NRES * 128 * 4);
  float* g2      = (float*)alloc((size_t)NRES * 128 * 4);
  float* b2      = (float*)alloc((size_t)NRES * 128 * 4);
  float* sgv     = (float*)alloc((size_t)NRES * 128 * 4);
  float* tgv     = (float*)alloc((size_t)NRES * 128 * 4);
  bf16*  fa      = (bf16*)alloc((size_t)NATOMS * 128 * 2);
  bf16*  qsb     = (bf16*)alloc((size_t)NATOMS * 128 * 2);
  bf16*  ksb     = (bf16*)alloc((size_t)NATOMS * 128 * 2);
  bf16*  vsb     = (bf16*)alloc((size_t)NATOMS * 128 * 2);
  float* sig_g   = (float*)alloc((size_t)NATOMS * 128 * 4);
  float* xbuf    = (float*)alloc((size_t)NATOMS * 128 * 4);
  bf16*  tbuf    = (bf16*)alloc((size_t)NATOMS * 128 * 2);
  bf16*  bmid    = (bf16*)alloc((size_t)NATOMS * 256 * 2);
  float* rawBig  = (float*)alloc((size_t)NATOMS * 512 * 4);
  // sequential-phase aliases inside rawBig
  float* rawR1 = rawBig;                 // 1024x512 (phase 1)
  float* rawR2 = rawBig + 1024 * 512;    // 1024x256 (phase 1)
  float* rawY  = rawBig;                 // 8192x128 (phase 4)

  auto packs = [&](const float* W, const float* sc, bf16* dst, int K, int Nsub, int stride, int colOff) {
    int tot = K * Nsub;
    pack_w_kernel<<<(tot + 255) / 256, 256, 0, stream>>>(W, sc, dst, K, Nsub, stride, colOff);
  };
  // fold LN elementwise weights into the conditioning GEMMs
  packs(a_gW, a_lnw, BcatA, 256, 128, 512, 0);
  packs(a_bW, a_lnw, BcatA, 256, 128, 512, 128);
  packs(t_gW, t_lnw, BcatA, 256, 128, 512, 256);
  packs(t_bW, t_lnw, BcatA, 256, 128, 512, 384);
  packs(sg_W, nullptr, BcatB, 256, 128, 256, 0);
  packs(tc_W, nullptr, BcatB, 256, 128, 256, 128);
  packs(q_W,  nullptr, Wqkvg, 128, 128, 512, 0);
  packs(kv_W, nullptr, Wqkvg, 128, 256, 512, 128);
  packs(g_W,  nullptr, Wqkvg, 128, 128, 512, 384);
  packs(l1_W, nullptr, Wlin12, 128, 256, 512, 0);
  packs(l2_W, nullptr, Wlin12, 128, 256, 512, 256);
  packs(lb_W, nullptr, Wlinb, 256, 128, 128, 0);

  // phase 1: residue conditioning
  res_prep_kernel<<<NRES / 8, 256, 0, stream>>>(resf, ln_res, res_raw);
  gemm_bf16_kernel<NRES, 512, 256><<<dim3(NRES / 64, 4), 256, 0, stream>>>(ln_res, BcatA, rawR1);
  gemm_bf16_kernel<NRES, 256, 256><<<dim3(NRES / 64, 2), 256, 0, stream>>>(res_raw, BcatB, rawR2);
  res_epilogue_kernel<<<NRES * 128 / 256, 256, 0, stream>>>(rawR1, rawR2, a_gb, t_gb, sg_b, tc_b,
                                                            g1, b1, g2, b2, sgv, tgv);
  // phase 2: adaLN + QKV/G projections
  adaln_kernel<<<NATOMS / 8, 256, 0, stream>>>(atomf, g1, b1, ridx, fa);
  gemm_bf16_kernel<NATOMS, 512, 128><<<dim3(NATOMS / 64, 4), 256, 0, stream>>>(fa, Wqkvg, rawBig);
  qkvg_epilogue_kernel<<<NATOMS * 128 / 256, 256, 0, stream>>>(rawBig, q_b, g_b, qsb, ksb, vsb, sig_g);
  // phase 3: block-local attention with pair bias, fused gating/residual
  attn_kernel<<<dim3(B_ * 128, H_), 256, 0, stream>>>(qsb, ksb, vsb, sig_g, sgv, atomf, pos,
                                                      rotm, transv, ridx, dv_W, d_W,
                                                      bij_lnw, bij_lnb, bij_W, xbuf);
  // phase 4: transition (adaLN -> gated MLP -> down-proj), fused final residual
  adaln_kernel<<<NATOMS / 8, 256, 0, stream>>>(xbuf, g2, b2, ridx, tbuf);
  gemm_bf16_kernel<NATOMS, 512, 128><<<dim3(NATOMS / 64, 4), 256, 0, stream>>>(tbuf, Wlin12, rawBig);
  bmid_epilogue_kernel<<<NATOMS * 256 / 256, 256, 0, stream>>>(rawBig, bmid);
  gemm_bf16_kernel<NATOMS, 128, 256><<<dim3(NATOMS / 64, 1), 256, 0, stream>>>(bmid, Wlinb, rawY);
  final_epilogue_kernel<<<NATOMS * 128 / 256, 256, 0, stream>>>(xbuf, rawY, tgv, ridx, (float*)d_out);
}
